// CnnMaskingAttention_54743653155401
// MI455X (gfx1250) — matrix operand order repaired and checked by reading, not compiled, not run on hardware
//
#include <hip/hip_runtime.h>
#include <math.h>

// ---- problem constants (from reference) ----
constexpr int SEQ   = 2048;
constexpr int NB    = 2;      // batch
constexpr int DM    = 1024;   // model dim
constexpr int NHEAD = 16;
constexpr int HDIM  = 64;     // dk == dv
constexpr int HDALL = NHEAD * HDIM;  // 1024
constexpr int BAND  = 128;    // SPAN

typedef __attribute__((ext_vector_type(16))) __bf16 v16bf;
typedef __attribute__((ext_vector_type(8)))  __bf16 v8bf;
typedef __attribute__((ext_vector_type(2)))  __bf16 v2bf;
typedef __attribute__((ext_vector_type(8)))  float  v8f;
typedef __attribute__((ext_vector_type(4)))  float  v4f;

// Concatenate two contiguous 8-element LDS runs into a WMMA 16-bit fragment.
// ISA 7.12.2: A-frag lane-half holds K = {khi..khi+7, 16+khi..16+khi+7}, so a
// fragment is exactly two b128 LDS loads when the runs are 16B-aligned.
#define FRAG16(lo, hi) __builtin_shufflevector((lo), (hi), \
    0, 1, 2, 3, 4, 5, 6, 7, 8, 9, 10, 11, 12, 13, 14, 15)

// ======================================================================
// Kernel 0: zero the dense attention output region (out-of-band == 0)
// ======================================================================
__global__ void zero_f4_kernel(float4* __restrict__ p, size_t n4) {
    size_t i = (size_t)blockIdx.x * blockDim.x + threadIdx.x;
    size_t stride = (size_t)gridDim.x * blockDim.x;
    float4 z = make_float4(0.f, 0.f, 0.f, 0.f);
    for (; i < n4; i += stride) p[i] = z;
}

// ======================================================================
// Kernel 1: C[M,N] = A[M,K] * W[K,N] + bias  (f32 in/out, bf16 WMMA core)
// Block = 256 threads (8 waves). Block tile 128x64, BK = 32.
// Double-buffered LDS + register pipeline: next tile's global loads are
// issued BEFORE the current tile's WMMAs; the loadcnt wait lands after them.
// ======================================================================
__global__ __launch_bounds__(256)
void gemm_bias_kernel(const float* __restrict__ A, const float* __restrict__ W,
                      const float* __restrict__ bias, float* __restrict__ C,
                      int M, int N, int Kd) {
    __shared__ __align__(16) __bf16 As[2][128][40];   // 2 x (128x32 + pad)
    __shared__ __align__(16) __bf16 BsT[2][64][40];   // 2 x transposed [n][k]

    const int tid  = threadIdx.x;
    const int lane = tid & 31;
    const int w    = tid >> 5;       // wave 0..7 -> M strip
    const int m0   = blockIdx.y * 128;
    const int n0   = blockIdx.x * 64;
    const int hi   = lane >> 4;      // lane-half
    const int nl   = lane & 15;
    const int khiA = hi * 8;
    const int khiB = hi * 8;

    float2 areg[8];                  // staged A pairs (in flight)
    float2 breg[4];                  // staged B pairs (in flight)
    v8f acc[4] = {v8f{}, v8f{}, v8f{}, v8f{}};

    auto load_tile = [&](int kk) {   // issue global loads only
        #pragma unroll
        for (int i = 0; i < 8; ++i) {
            int e = tid + i * 256;               // pair index, 0..2047
            int r = e >> 4, cp = e & 15;
            areg[i] = *(const float2*)&A[(size_t)(m0 + r) * Kd + kk + 2 * cp];
        }
        #pragma unroll
        for (int i = 0; i < 4; ++i) {
            int e = tid + i * 256;               // pair index, 0..1023
            int r = e >> 5, np = e & 31;
            breg[i] = *(const float2*)&W[(size_t)(kk + r) * N + n0 + 2 * np];
        }
    };
    auto store_tile = [&](int buf) { // convert + LDS store
        #pragma unroll
        for (int i = 0; i < 8; ++i) {
            int e = tid + i * 256;
            int r = e >> 4, cp = e & 15;
            v2bf p = {(__bf16)areg[i].x, (__bf16)areg[i].y};
            *(v2bf*)&As[buf][r][2 * cp] = p;
        }
        #pragma unroll
        for (int i = 0; i < 4; ++i) {
            int e = tid + i * 256;
            int r = e >> 5, np = e & 31;
            BsT[buf][2 * np][r]     = (__bf16)breg[i].x;
            BsT[buf][2 * np + 1][r] = (__bf16)breg[i].y;
        }
    };

    load_tile(0);
    store_tile(0);
    __syncthreads();

    for (int kk = 0; kk < Kd; kk += 32) {
        const int  cur  = (kk >> 5) & 1;
        const bool more = (kk + 32) < Kd;
        if (more) load_tile(kk + 32);            // loads in flight over WMMAs
        if (kk + 64 < Kd)                        // prefetch tile after next
            __builtin_prefetch(&A[(size_t)(m0 + (tid >> 1)) * Kd + kk + 64], 0, 1);

        const __bf16* arow = &As[cur][w * 16 + nl][0];
        v16bf af = FRAG16(*(const v8bf*)(arow + khiA),
                          *(const v8bf*)(arow + 16 + khiA));
        #pragma unroll
        for (int j = 0; j < 4; ++j) {
            const __bf16* brow = &BsT[cur][j * 16 + nl][0];
            v16bf bf = FRAG16(*(const v8bf*)(brow + khiB),
                              *(const v8bf*)(brow + 16 + khiB));
            acc[j] = __builtin_amdgcn_wmma_f32_16x16x32_bf16(false, af, false, bf,
                                                             (short)0, acc[j], false, false);
        }

        if (more) store_tile(cur ^ 1);           // waits for loads here
        __syncthreads();
    }

    // C/D layout: VGPR r -> row (r + 8*lane_half), col = lane%16
    #pragma unroll
    for (int j = 0; j < 4; ++j) {
        #pragma unroll
        for (int r = 0; r < 8; ++r) {
            int gm = m0 + w * 16 + r + hi * 8;
            int gn = n0 + j * 16 + nl;
            C[(size_t)gm * N + gn] = acc[j][r] + bias[gn];
        }
    }
}

// ======================================================================
// Kernel 2: banded attention. One block = 16 query rows of one (h,b).
// Block = 128 threads (4 waves). Band = 17 key tiles of 16 (272 cols).
// ======================================================================
__global__ __launch_bounds__(128)
void banded_attn_kernel(const float* __restrict__ Qm, const float* __restrict__ Km,
                        const float* __restrict__ Vm, float* __restrict__ attn,
                        float* __restrict__ ctx) {
    __shared__ __align__(16) __bf16 Qs[16][72];                 // pre-scaled
    __shared__ __align__(16) unsigned char KVraw[272 * 72 * 2]; // 39168B union
    __shared__ __align__(16) float Ss[16][292];                 // 1168B rows

    __bf16 (*KVs)[72]  = (__bf16 (*)[72])KVraw;    // K band: [jj][k]
    __bf16 (*VsT)[296] = (__bf16 (*)[296])KVraw;   // V transposed: [v][jj]

    const int tid  = threadIdx.x;
    const int lane = tid & 31;
    const int wv   = tid >> 5;
    const int qt   = blockIdx.x;
    const int hb   = blockIdx.y;
    const int h    = hb / NB;
    const int b    = hb % NB;
    const int i0   = qt * 16;
    const int jbase = i0 - BAND;
    const int hi   = lane >> 4;
    const int nl   = lane & 15;
    const int khiA = hi * 8, khiB = hi * 8;
    const bool noclip = (jbase >= 0) && (jbase + 272 <= SEQ);

    const float* Kbase = &Km[((size_t)0 * NB + b) * HDALL + h * HDIM];
    const float* Vbase = &Vm[((size_t)0 * NB + b) * HDALL + h * HDIM];

    // ---- phase 1: stage Q (scaled, batched) and K band ----
    {
        float2 t4[4];
        #pragma unroll
        for (int i = 0; i < 4; ++i) {
            int e = tid + i * 128;               // pair index, 0..511
            int r = e >> 5, cp = e & 31;
            t4[i] = *(const float2*)&Qm[((size_t)(i0 + r) * NB + b) * HDALL + h * HDIM + 2 * cp];
        }
        #pragma unroll
        for (int i = 0; i < 4; ++i) {
            int e = tid + i * 128;
            int r = e >> 5, cp = e & 31;
            v2bf p = {(__bf16)(t4[i].x * 0.125f), (__bf16)(t4[i].y * 0.125f)};
            *(v2bf*)&Qs[r][2 * cp] = p;
        }
    }
    if (noclip) {
        for (int i = 0; i < 68; i += 4) {        // 272x64 = 8704 pairs, batch 4
            float2 t4[4];
            #pragma unroll
            for (int u = 0; u < 4; ++u) {
                int e = tid + (i + u) * 128;
                int jj = e >> 5, cp = e & 31;
                t4[u] = *(const float2*)&Kbase[(size_t)(jbase + jj) * NB * HDALL + 2 * cp];
            }
            #pragma unroll
            for (int u = 0; u < 4; ++u) {
                int e = tid + (i + u) * 128;
                int jj = e >> 5, cp = e & 31;
                v2bf p = {(__bf16)t4[u].x, (__bf16)t4[u].y};
                *(v2bf*)&KVs[jj][2 * cp] = p;
            }
        }
    } else {
        for (int i = 0; i < 68; ++i) {
            int e = tid + i * 128;
            int jj = e >> 5, cp = e & 31;
            int j = jbase + jj;
            float kx = 0.f, ky = 0.f;
            if (j >= 0 && j < SEQ) {
                const float2 v = *(const float2*)&Kbase[(size_t)j * NB * HDALL + 2 * cp];
                kx = v.x; ky = v.y;
            }
            v2bf p = {(__bf16)kx, (__bf16)ky};
            *(v2bf*)&KVs[jj][2 * cp] = p;
        }
    }
    __syncthreads();

    // ---- phase 2: S = (Q/sqrt(dk)) * K^T over 17 tiles (wave-uniform) ----
    const __bf16* qrow = &Qs[nl][0];
    v16bf aq0 = FRAG16(*(const v8bf*)(qrow + khiA),      *(const v8bf*)(qrow + 16 + khiA));
    v16bf aq1 = FRAG16(*(const v8bf*)(qrow + 32 + khiA), *(const v8bf*)(qrow + 48 + khiA));

    for (int t = wv; t < 17; t += 4) {
        const __bf16* krow = &KVs[t * 16 + nl][0];       // B[k][n] = K[j=n][k]
        v16bf b0 = FRAG16(*(const v8bf*)(krow + khiB),      *(const v8bf*)(krow + 16 + khiB));
        v16bf b1 = FRAG16(*(const v8bf*)(krow + 32 + khiB), *(const v8bf*)(krow + 48 + khiB));
        v8f acc = {};
        acc = __builtin_amdgcn_wmma_f32_16x16x32_bf16(false, aq0, false, b0,
                                                      (short)0, acc, false, false);
        acc = __builtin_amdgcn_wmma_f32_16x16x32_bf16(false, aq1, false, b1,
                                                      (short)0, acc, false, false);
        #pragma unroll
        for (int r = 0; r < 8; ++r) {
            int rl = r + hi * 8;
            int i  = i0 + rl;
            int jj = t * 16 + nl;
            int j  = jbase + jj;
            bool ok = (j >= 0) && (j < SEQ) && (j >= i - BAND) && (j <= i + BAND);
            Ss[rl][jj] = ok ? acc[r] : -__builtin_inff();
        }
    }
    __syncthreads();

    // ---- phase 3: softmax over 272 cols (8 lanes per row), probs -> Ss ----
    {
        int row = tid >> 3, tg = tid & 7;
        float m = -__builtin_inff();
        for (int jj = tg; jj < 272; jj += 8) m = fmaxf(m, Ss[row][jj]);
        m = fmaxf(m, __shfl_xor(m, 1, 8));
        m = fmaxf(m, __shfl_xor(m, 2, 8));
        m = fmaxf(m, __shfl_xor(m, 4, 8));
        float s = 0.f;
        for (int jj = tg; jj < 272; jj += 8) {
            float e = __expf(Ss[row][jj] - m);
            Ss[row][jj] = e;
            s += e;
        }
        s += __shfl_xor(s, 1, 8);
        s += __shfl_xor(s, 2, 8);
        s += __shfl_xor(s, 4, 8);
        float inv = 1.f / s;
        for (int jj = tg; jj < 272; jj += 8) Ss[row][jj] *= inv;
        Ss[row][272 + tg] = 0.f;   // zero pad cols 272..287 for phase 4
        Ss[row][280 + tg] = 0.f;
    }
    __syncthreads();

    // ---- phase 3b: coalesced attn scatter + stage V transposed ----
    {
        size_t abase = (((size_t)h * NB + b) * SEQ + i0) * (size_t)SEQ;
        for (int i = 0; i < 36; i += 6) {        // 16 rows x 288 cols, batch 6
            float pv[6];
            #pragma unroll
            for (int u = 0; u < 6; ++u) {
                int e = tid + (i + u) * 128;
                pv[u] = Ss[e / 288][e % 288];
            }
            #pragma unroll
            for (int u = 0; u < 6; ++u) {
                int e = tid + (i + u) * 128;
                int row = e / 288, jj = e % 288;
                int j = jbase + jj;
                if (jj < 272 && j >= 0 && j < SEQ)
                    attn[abase + (size_t)row * SEQ + j] = pv[u];
            }
        }
    }
    if (noclip) {
        for (int i = 0; i < 68; i += 4) {        // V band -> VsT[v][jj], batch 4
            float2 t4[4];
            #pragma unroll
            for (int u = 0; u < 4; ++u) {
                int e = tid + (i + u) * 128;
                int jj = e >> 5, cp = e & 31;
                t4[u] = *(const float2*)&Vbase[(size_t)(jbase + jj) * NB * HDALL + 2 * cp];
            }
            #pragma unroll
            for (int u = 0; u < 4; ++u) {
                int e = tid + (i + u) * 128;
                int jj = e >> 5, cp = e & 31;
                VsT[2 * cp][jj]     = (__bf16)t4[u].x;
                VsT[2 * cp + 1][jj] = (__bf16)t4[u].y;
            }
        }
    } else {
        for (int i = 0; i < 68; ++i) {
            int e = tid + i * 128;
            int jj = e >> 5, cp = e & 31;
            int j = jbase + jj;
            float vx = 0.f, vy = 0.f;
            if (j >= 0 && j < SEQ) {
                const float2 v = *(const float2*)&Vbase[(size_t)j * NB * HDALL + 2 * cp];
                vx = v.x; vy = v.y;
            }
            VsT[2 * cp][jj]     = (__bf16)vx;
            VsT[2 * cp + 1][jj] = (__bf16)vy;
        }
    }
    #pragma unroll
    for (int i = 0; i < 8; ++i) {                // zero pad VsT cols 272..287
        int e = tid + i * 128;
        int v = e >> 4, jj = 272 + (e & 15);
        VsT[v][jj] = (__bf16)0.f;
    }
    __syncthreads();

    // ---- phase 4: O = P(16x288) * V(288x64); wave wv owns 16 v-columns ----
    {
        v8f acc = {};
        const int n0v = wv * 16;
        const float*  srow = &Ss[nl][0];            // P: query row = lane%16
        const __bf16* vrow = &VsT[n0v + nl][0];     // B[k][n] = V[j=k][v=n]
        for (int kc = 0; kc < 9; ++kc) {
            v4f s0 = *(const v4f*)(srow + kc * 32 + khiA);
            v4f s1 = *(const v4f*)(srow + kc * 32 + khiA + 4);
            v4f s2 = *(const v4f*)(srow + kc * 32 + 16 + khiA);
            v4f s3 = *(const v4f*)(srow + kc * 32 + 16 + khiA + 4);
            v16bf ap;
            #pragma unroll
            for (int q = 0; q < 4; ++q) {
                ap[q]      = (__bf16)s0[q];
                ap[4 + q]  = (__bf16)s1[q];
                ap[8 + q]  = (__bf16)s2[q];
                ap[12 + q] = (__bf16)s3[q];
            }
            v16bf bvf = FRAG16(*(const v8bf*)(vrow + kc * 32 + khiB),
                               *(const v8bf*)(vrow + kc * 32 + 16 + khiB));
            acc = __builtin_amdgcn_wmma_f32_16x16x32_bf16(false, ap, false, bvf,
                                                          (short)0, acc, false, false);
        }
        #pragma unroll
        for (int r = 0; r < 8; ++r) {
            int rl = r + hi * 8;
            int i  = i0 + rl;
            ctx[((size_t)i * NB + b) * HDALL + h * HDIM + n0v + nl] = acc[r];
        }
    }
}

// ======================================================================
// Kernel 3: y = LayerNorm(fc + residual) * gamma + beta   (one row/block)
// ======================================================================
__global__ __launch_bounds__(256)
void ln_residual_kernel(const float* __restrict__ fc, const float* __restrict__ resid,
                        const float* __restrict__ gamma, const float* __restrict__ beta,
                        float* __restrict__ y) {
    __shared__ float red[16];
    const int row = blockIdx.x;
    const int tid = threadIdx.x;
    const int c0  = tid * 4;

    v4f xv = *(const v4f*)&fc[(size_t)row * DM + c0];
    v4f rv = *(const v4f*)&resid[(size_t)row * DM + c0];
    xv = xv + rv;
    float s  = xv[0] + xv[1] + xv[2] + xv[3];
    float sq = xv[0] * xv[0] + xv[1] * xv[1] + xv[2] * xv[2] + xv[3] * xv[3];

    for (int off = 16; off > 0; off >>= 1) {
        s  += __shfl_xor(s,  off, 32);
        sq += __shfl_xor(sq, off, 32);
    }
    int wv = tid >> 5, lane = tid & 31;
    if (lane == 0) { red[wv] = s; red[8 + wv] = sq; }
    __syncthreads();
    if (wv == 0) {
        float ps = (lane < 8) ? red[lane] : 0.f;
        float pq = (lane < 8) ? red[8 + lane] : 0.f;
        for (int off = 4; off > 0; off >>= 1) {
            ps += __shfl_xor(ps, off, 32);
            pq += __shfl_xor(pq, off, 32);
        }
        if (lane == 0) { red[0] = ps; red[8] = pq; }
    }
    __syncthreads();
    float mu  = red[0] * (1.f / DM);
    float var = red[8] * (1.f / DM) - mu * mu;
    float rs  = rsqrtf(var + 1e-5f);

    v4f gv  = *(const v4f*)&gamma[c0];
    v4f bvv = *(const v4f*)&beta[c0];
    v4f out;
    #pragma unroll
    for (int q = 0; q < 4; ++q) out[q] = (xv[q] - mu) * rs * gv[q] + bvv[q];
    *(v4f*)&y[(size_t)row * DM + c0] = out;
}

// ======================================================================
extern "C" void kernel_launch(void* const* d_in, const int* in_sizes, int n_in,
                              void* d_out, int out_size, void* d_ws, size_t ws_size,
                              hipStream_t stream) {
    const float* query = (const float*)d_in[0];
    const float* key   = (const float*)d_in[1];
    const float* value = (const float*)d_in[2];
    const float* Wq    = (const float*)d_in[3];
    const float* bq    = (const float*)d_in[4];
    const float* Wk    = (const float*)d_in[5];
    const float* bk    = (const float*)d_in[6];
    const float* Wv    = (const float*)d_in[7];
    const float* bv    = (const float*)d_in[8];
    const float* Wfc   = (const float*)d_in[9];
    const float* bfc   = (const float*)d_in[10];
    const float* gamma = (const float*)d_in[11];
    const float* beta  = (const float*)d_in[12];

    float* out_y    = (float*)d_out;                       // (L,B,D)
    float* out_attn = out_y + (size_t)SEQ * NB * DM;       // (H,B,L,L)

    const size_t MAT = (size_t)SEQ * NB * HDALL;           // 4096*1024 floats
    float* Qb  = (float*)d_ws;
    float* Kb  = Qb + MAT;
    float* Vb  = Kb + MAT;
    float* Ctx = Vb + MAT;
    float* Fc  = Ctx + MAT;

    // 0) clear dense attn output (softmax outside the band == 0)
    {
        size_t n4 = ((size_t)NHEAD * NB * SEQ * SEQ) / 4;
        zero_f4_kernel<<<4096, 256, 0, stream>>>((float4*)out_attn, n4);
    }

    // 1) Q/K/V projections (M=4096, N=1024, K=1024)
    dim3 gg(HDALL / 64, (SEQ * NB) / 128);
    gemm_bias_kernel<<<gg, 256, 0, stream>>>(query, Wq, bq, Qb, SEQ * NB, HDALL, DM);
    gemm_bias_kernel<<<gg, 256, 0, stream>>>(key,   Wk, bk, Kb, SEQ * NB, HDALL, DM);
    gemm_bias_kernel<<<gg, 256, 0, stream>>>(value, Wv, bv, Vb, SEQ * NB, HDALL, DM);

    // 2) banded attention: (L/16) query tiles x (H*B)
    banded_attn_kernel<<<dim3(SEQ / 16, NHEAD * NB), 128, 0, stream>>>(
        Qb, Kb, Vb, out_attn, Ctx);

    // 3) output FC
    gemm_bias_kernel<<<gg, 256, 0, stream>>>(Ctx, Wfc, bfc, Fc, SEQ * NB, DM, HDALL);

    // 4) residual + LayerNorm
    ln_residual_kernel<<<SEQ * NB, 256, 0, stream>>>(Fc, query, gamma, beta, out_y);
}
